// IDRNetwork_37675453120946
// MI455X (gfx1250) — matrix operand order repaired and checked by reading, not compiled, not run on hardware
//
#include <hip/hip_runtime.h>
#include <math.h>

// ---------------------------------------------------------------------------
// Types for CDNA5 WMMA (wave32): v_wmma_f32_16x16x32_f16
// ---------------------------------------------------------------------------
typedef __attribute__((ext_vector_type(16))) _Float16 v16h;
typedef __attribute__((ext_vector_type(8)))  _Float16 v8h;
typedef __attribute__((ext_vector_type(8)))  float    v8f;

#define NPTS 32768
#define INV_SQRT2 0.70710678118654752440f
#define SBS 56   // LDS B-tile row stride in f16 (112B = 28 dwords: conflict-free, 16B aligned)
#define SBUF (64 * SBS)   // one B tile buffer (f16 elements)

// ---------------------------------------------------------------------------
// Device helpers (branch-free, hardware-transcendental activations)
// ---------------------------------------------------------------------------
__device__ __forceinline__ float softplus100f(float v) {
    // logaddexp(0, 100*v)/100 = (max(z,0) + log(1+exp(-|z|)))/100.
    // When exp(-|z|) underflows 1+t, the dropped term is < f32 eps relative
    // to max(z,0), so plain __logf(1+t) is accurate without log1pf branches.
    float z = 100.0f * v;
    float t = __expf(-fabsf(z));
    return (fmaxf(z, 0.0f) + __logf(1.0f + t)) * 0.01f;
}

__device__ __forceinline__ float fast_tanhf(float v) {
    // tanh(x) = 1 - 2/(exp(2x)+1); exp overflow -> +inf -> 1, underflow -> -1.
    float t = __expf(2.0f * v);
    return 1.0f - 2.0f / (t + 1.0f);
}

// ---------------------------------------------------------------------------
// Embedding: points [N,3] -> f16 inp [N,64] (cols 0..38 valid, 39..63 zero)
// ---------------------------------------------------------------------------
__global__ void embed_kernel(const float* __restrict__ pts,
                             _Float16* __restrict__ inp, int n) {
    int r = blockIdx.x * blockDim.x + threadIdx.x;
    if (r >= n) return;
    float p0 = pts[3 * r + 0], p1 = pts[3 * r + 1], p2 = pts[3 * r + 2];
    _Float16* row = inp + (size_t)r * 64;
    row[0] = (_Float16)p0; row[1] = (_Float16)p1; row[2] = (_Float16)p2;
    float f = 1.0f;
    #pragma unroll
    for (int i = 0; i < 6; ++i) {
        float a0 = f * p0, a1 = f * p1, a2 = f * p2;   // |arg| <= 32
        row[3 + 6 * i + 0] = (_Float16)__sinf(a0);
        row[3 + 6 * i + 1] = (_Float16)__sinf(a1);
        row[3 + 6 * i + 2] = (_Float16)__sinf(a2);
        row[3 + 6 * i + 3] = (_Float16)__cosf(a0);
        row[3 + 6 * i + 4] = (_Float16)__cosf(a1);
        row[3 + 6 * i + 5] = (_Float16)__cosf(a2);
        f *= 2.0f;
    }
    #pragma unroll
    for (int j = 39; j < 64; ++j) row[j] = (_Float16)0.0f;
}

// Fill skip-concat region: buf[:, 473..511] = inp[:, 0..38] * 1/sqrt(2)
__global__ void fill_skip_kernel(const _Float16* __restrict__ inp,
                                 _Float16* __restrict__ buf, int n) {
    int r = blockIdx.x * blockDim.x + threadIdx.x;
    if (r >= n) return;
    const _Float16* irow = inp + (size_t)r * 64;
    _Float16* orow = buf + (size_t)r * 512;
    #pragma unroll
    for (int j = 0; j < 39; ++j)
        orow[473 + j] = (_Float16)((float)irow[j] * INV_SQRT2);
}

// ---------------------------------------------------------------------------
// Weight conversion: f32 [B,O,K] row-major -> f16 [B,Opad,Kpad] zero-padded
// ---------------------------------------------------------------------------
__global__ void convert_w_kernel(const float* __restrict__ src,
                                 _Float16* __restrict__ dst,
                                 int B, int O, int K, int Opad, int Kpad) {
    long long idx = (long long)blockIdx.x * blockDim.x + threadIdx.x;
    long long total = (long long)B * Opad * Kpad;
    if (idx >= total) return;
    int k = (int)(idx % Kpad);
    long long t = idx / Kpad;
    int o = (int)(t % Opad);
    int b = (int)(t / Opad);
    float v = (o < O && k < K) ? src[((long long)b * O + o) * K + k] : 0.0f;
    dst[idx] = (_Float16)v;
}

__global__ void convert_b_kernel(const float* __restrict__ src,
                                 float* __restrict__ dst,
                                 int B, int O, int Opad) {
    int idx = blockIdx.x * blockDim.x + threadIdx.x;
    if (idx >= B * Opad) return;
    int o = idx % Opad;
    int b = idx / Opad;
    dst[idx] = (o < O) ? src[b * O + o] : 0.0f;
}

// ---------------------------------------------------------------------------
// Tiled WMMA GEMM:  Out = act( X[M,Kpad] * W[Opad,Kpad]^T + bias ) * scale
//  - 256 threads = 8 waves; block tile 256(M) x 64(N); wave tile 32 x 64
//  - B tile (64x32 f16) double-buffered in LDS (one barrier per k-step),
//    staged via register pipeline; B fragments software-pipelined across nt.
//  - A fragments loaded straight from global (no intra-block redundancy).
//  - ACT: 0=none 1=softplus100 2=relu 3=tanh
//  - OUTF32: store f32 to d_out (row stride ldo, column colOff) else f16
//  - cluster mode: weight slab selected by block_row / clusterRows
// ---------------------------------------------------------------------------
template <int ACT, bool OUTF32>
__global__ __launch_bounds__(256) void gemm_wmma_kernel(
    const _Float16* __restrict__ X, int ldx,
    const _Float16* __restrict__ W, int ldw, long long wstride, int clusterRows,
    const float* __restrict__ bias, int bstride,
    void* __restrict__ Out, int ldo, int colOff, int Ovalid, int Opad,
    int Kpad, float scale) {
    __shared__ _Float16 sB[2 * SBUF];

    const int tid  = threadIdx.x;
    const int wave = tid >> 5;
    const int lane = tid & 31;
    const int lr   = lane & 15;   // row/col within a 16-tile
    const int lh   = lane >> 4;   // lane half

    const int m_blk  = blockIdx.x * 256;
    const int n_base = blockIdx.y * 64;

    const int cl = m_blk / clusterRows;
    const _Float16* Wb = W + (long long)cl * wstride;
    const float*    bb = bias + cl * bstride;

    // Cooperative B staging: thread t covers row t>>2, cols (t&3)*8 .. +8
    const int srow = tid >> 2;
    int wsr = n_base + srow;
    if (wsr >= Opad) wsr = Opad - 1;          // clamp: in-bounds reads, guarded stores
    const _Float16* wstage = Wb + (size_t)wsr * ldw + (tid & 3) * 8;
    const int soff = srow * SBS + (tid & 3) * 8;

    const int m_base = m_blk + wave * 32;
    const _Float16* xrow0 = X + (size_t)(m_base + lr) * ldx;
    const _Float16* xrow1 = X + (size_t)(m_base + 16 + lr) * ldx;

    v8f acc[2][4] = {};

    const int nk = Kpad >> 5;

    // ---- staging prologue: chunk 0 -> buf0; prefetch chunk 1 into regs ----
    v8h breg = *(const v8h*)(wstage);
    *(v8h*)(sB + soff) = breg;
    if (nk > 1) breg = *(const v8h*)(wstage + 32);
    __syncthreads();

    #pragma unroll 2
    for (int kk = 0; kk < nk; ++kk) {
        const int k0  = kk << 5;
        const int cur = (kk & 1) * SBUF;
        const int nxt = ((kk + 1) & 1) * SBUF;

        __builtin_prefetch(xrow0 + k0 + 64, 0, 1);   // global_prefetch_b8

        // A fragments (16x32 f16 each): lane=M row; halves per ISA 7.12.2
        v8h a0lo = *(const v8h*)(xrow0 + k0 + lh * 8);
        v8h a0hi = *(const v8h*)(xrow0 + k0 + 16 + lh * 8);
        v16h a0 = __builtin_shufflevector(a0lo, a0hi,
            0,1,2,3,4,5,6,7,8,9,10,11,12,13,14,15);
        v8h a1lo = *(const v8h*)(xrow1 + k0 + lh * 8);
        v8h a1hi = *(const v8h*)(xrow1 + k0 + 16 + lh * 8);
        v16h a1 = __builtin_shufflevector(a1lo, a1hi,
            0,1,2,3,4,5,6,7,8,9,10,11,12,13,14,15);

        // B fragments software-pipelined over nt: loads for nt+1 are issued
        // before the WMMAs consuming nt, so waits retire at dscnt<=2.
        const _Float16* sbase = sB + cur + lr * SBS + lh * 8;
        v8h c0 = *(const v8h*)(sbase);
        v8h c1 = *(const v8h*)(sbase + 16);
        v16h bcur = __builtin_shufflevector(c0, c1,
            0,1,2,3,4,5,6,7,8,9,10,11,12,13,14,15);

        #pragma unroll
        for (int nt = 0; nt < 4; ++nt) {
            v16h bnext = bcur;
            if (nt < 3) {
                const _Float16* sp = sbase + (nt + 1) * 16 * SBS;
                v8h nlo = *(const v8h*)(sp);
                v8h nhi = *(const v8h*)(sp + 16);
                bnext = __builtin_shufflevector(nlo, nhi,
                    0,1,2,3,4,5,6,7,8,9,10,11,12,13,14,15);
            }
            acc[0][nt] = __builtin_amdgcn_wmma_f32_16x16x32_f16(
                false, a0, false, bcur, (short)0, acc[0][nt], false, false);
            acc[1][nt] = __builtin_amdgcn_wmma_f32_16x16x32_f16(
                false, a1, false, bcur, (short)0, acc[1][nt], false, false);
            bcur = bnext;
        }

        // stage chunk kk+1 into the other buffer; prefetch chunk kk+2
        if (kk + 1 < nk) {
            *(v8h*)(sB + nxt + soff) = breg;
            if (kk + 2 < nk)
                breg = *(const v8h*)(wstage + k0 + 64);
            __syncthreads();   // single barrier per k-step (ping-pong buffers)
        }
    }

    // Epilogue: bias + activation + store (C layout: row = base + lh*8 + i)
    #pragma unroll
    for (int mt = 0; mt < 2; ++mt) {
        #pragma unroll
        for (int nt = 0; nt < 4; ++nt) {
            const int col = n_base + nt * 16 + lr;
            if (col >= Ovalid) continue;
            const float bv = bb[col];
            #pragma unroll
            for (int i = 0; i < 8; ++i) {
                float v = acc[mt][nt][i] + bv;
                if (ACT == 1) v = softplus100f(v);
                else if (ACT == 2) v = fmaxf(v, 0.0f);
                else if (ACT == 3) v = fast_tanhf(v);
                v *= scale;
                const int row = m_base + mt * 16 + lh * 8 + i;
                if (OUTF32)
                    ((float*)Out)[(size_t)row * ldo + colOff + col] = v;
                else
                    ((_Float16*)Out)[(size_t)row * ldo + col] = (_Float16)v;
            }
        }
    }
}

// ---------------------------------------------------------------------------
// Host side
// ---------------------------------------------------------------------------
static void launch_gemm(int act, bool outf32,
                        const _Float16* X, int ldx,
                        const _Float16* W, int ldw, long long wstride, int clusterRows,
                        const float* bias, int bstride,
                        void* out, int ldo, int colOff, int Ovalid, int Opad, int Kpad,
                        float scale, hipStream_t stream) {
    dim3 grid(NPTS / 256, (Opad + 63) / 64), block(256);
#define GL(A, F) gemm_wmma_kernel<A, F><<<grid, block, 0, stream>>>( \
        X, ldx, W, ldw, wstride, clusterRows, bias, bstride,          \
        out, ldo, colOff, Ovalid, Opad, Kpad, scale)
    if (outf32) {
        if (act == 3) GL(3, true);
        else if (act == 1) GL(1, true);
        else GL(0, true);
    } else {
        if (act == 1) GL(1, false);
        else if (act == 2) GL(2, false);
        else GL(0, false);
    }
#undef GL
}

struct LW {
    int batch, O, K, Opad, Kpad;
    const float* w; const float* b;
    _Float16* wd; float* bd;
};

extern "C" void kernel_launch(void* const* d_in, const int* in_sizes, int n_in,
                              void* d_out, int out_size, void* d_ws, size_t ws_size,
                              hipStream_t stream) {
    (void)in_sizes; (void)n_in; (void)out_size; (void)ws_size;
    const int N = NPTS;
    const float* points = (const float*)d_in[0];
    float* out = (float*)d_out;

    // ---- workspace carve-up ----
    char* wsb = (char*)d_ws;
    size_t cur = 0;
    auto take = [&](size_t bytes) -> void* {
        void* p = wsb + cur;
        cur += bytes;
        cur = (cur + 255) & ~(size_t)255;
        return p;
    };
    _Float16* inp  = (_Float16*)take((size_t)N * 64 * sizeof(_Float16));
    _Float16* bufA = (_Float16*)take((size_t)N * 512 * sizeof(_Float16));
    _Float16* bufB = (_Float16*)take((size_t)N * 512 * sizeof(_Float16));

    // ---- layer descriptors ----
    static const int implO[9] = {512, 512, 512, 473, 512, 512, 512, 512, 257};
    static const int dispO[4] = {512, 512, 512, 3};
    static const int multO[3] = {128, 128, 3};
    static const int multK[3] = {39, 128, 128};

    LW ly[16];
    int nl = 0;
    for (int l = 0; l < 9; ++l) {
        LW L; L.batch = 1; L.O = implO[l]; L.K = (l == 0) ? 39 : 512;
        L.Opad = (L.O + 15) & ~15; L.Kpad = (L.K + 31) & ~31;
        L.w = (const float*)d_in[1 + 2 * l]; L.b = (const float*)d_in[2 + 2 * l];
        ly[nl++] = L;
    }
    for (int l = 0; l < 4; ++l) {
        LW L; L.batch = 1; L.O = dispO[l]; L.K = (l == 0) ? 39 : 512;
        L.Opad = (L.O + 15) & ~15; L.Kpad = (L.K + 31) & ~31;
        L.w = (const float*)d_in[19 + 2 * l]; L.b = (const float*)d_in[20 + 2 * l];
        ly[nl++] = L;
    }
    for (int l = 0; l < 3; ++l) {
        LW L; L.batch = 64; L.O = multO[l]; L.K = multK[l];
        L.Opad = (L.O + 15) & ~15; L.Kpad = (L.K + 31) & ~31;
        L.w = (const float*)d_in[27 + 2 * l]; L.b = (const float*)d_in[28 + 2 * l];
        ly[nl++] = L;
    }

    for (int i = 0; i < nl; ++i) {
        LW& L = ly[i];
        L.wd = (_Float16*)take((size_t)L.batch * L.Opad * L.Kpad * sizeof(_Float16));
        L.bd = (float*)take((size_t)L.batch * L.Opad * sizeof(float));
    }

    // ---- weight / bias conversion to padded f16 / f32 ----
    for (int i = 0; i < nl; ++i) {
        LW& L = ly[i];
        long long totW = (long long)L.batch * L.Opad * L.Kpad;
        convert_w_kernel<<<(unsigned)((totW + 255) / 256), 256, 0, stream>>>(
            L.w, L.wd, L.batch, L.O, L.K, L.Opad, L.Kpad);
        int totB = L.batch * L.Opad;
        convert_b_kernel<<<(totB + 255) / 256, 256, 0, stream>>>(
            L.b, L.bd, L.batch, L.O, L.Opad);
    }

    // ---- embedding ----
    embed_kernel<<<N / 256, 256, 0, stream>>>(points, inp, N);

    const int BIG = 1 << 30;  // non-cluster mode

    // ---- ImplicitNetwork (softplus beta=100, skip at layer 4) ----
    launch_gemm(1, false, inp, 64, ly[0].wd, ly[0].Kpad, 0, BIG, ly[0].bd, 0,
                bufA, 512, 0, ly[0].O, ly[0].Opad, ly[0].Kpad, 1.0f, stream);
    launch_gemm(1, false, bufA, 512, ly[1].wd, 512, 0, BIG, ly[1].bd, 0,
                bufB, 512, 0, 512, 512, 512, 1.0f, stream);
    launch_gemm(1, false, bufB, 512, ly[2].wd, 512, 0, BIG, ly[2].bd, 0,
                bufA, 512, 0, 512, 512, 512, 1.0f, stream);
    // L3: -> bufB cols 0..472, scaled by 1/sqrt(2) (skip-concat fold)
    launch_gemm(1, false, bufA, 512, ly[3].wd, 512, 0, BIG, ly[3].bd, 0,
                bufB, 512, 0, ly[3].O, ly[3].Opad, 512, INV_SQRT2, stream);
    fill_skip_kernel<<<N / 256, 256, 0, stream>>>(inp, bufB, N);
    launch_gemm(1, false, bufB, 512, ly[4].wd, 512, 0, BIG, ly[4].bd, 0,
                bufA, 512, 0, 512, 512, 512, 1.0f, stream);
    launch_gemm(1, false, bufA, 512, ly[5].wd, 512, 0, BIG, ly[5].bd, 0,
                bufB, 512, 0, 512, 512, 512, 1.0f, stream);
    launch_gemm(1, false, bufB, 512, ly[6].wd, 512, 0, BIG, ly[6].bd, 0,
                bufA, 512, 0, 512, 512, 512, 1.0f, stream);
    launch_gemm(1, false, bufA, 512, ly[7].wd, 512, 0, BIG, ly[7].bd, 0,
                bufB, 512, 0, 512, 512, 512, 1.0f, stream);
    // L8: no activation, f32 -> d_out cols 0..256
    launch_gemm(0, true, bufB, 512, ly[8].wd, 512, 0, BIG, ly[8].bd, 0,
                out, 263, 0, ly[8].O, ly[8].Opad, 512, 1.0f, stream);

    // ---- DisplacementNetwork (relu x3, tanh out) ----
    launch_gemm(2, false, inp, 64, ly[9].wd, ly[9].Kpad, 0, BIG, ly[9].bd, 0,
                bufA, 512, 0, 512, 512, ly[9].Kpad, 1.0f, stream);
    launch_gemm(2, false, bufA, 512, ly[10].wd, 512, 0, BIG, ly[10].bd, 0,
                bufB, 512, 0, 512, 512, 512, 1.0f, stream);
    launch_gemm(2, false, bufB, 512, ly[11].wd, 512, 0, BIG, ly[11].bd, 0,
                bufA, 512, 0, 512, 512, 512, 1.0f, stream);
    // D3: tanh, f32 -> d_out cols 257..259
    launch_gemm(3, true, bufA, 512, ly[12].wd, 512, 0, BIG, ly[12].bd, 0,
                out, 263, 257, ly[12].O, ly[12].Opad, 512, 1.0f, stream);

    // ---- MultiDisplacementNetwork (64 clusters of 512 rows) ----
    launch_gemm(2, false, inp, 64, ly[13].wd, ly[13].Kpad,
                (long long)ly[13].Opad * ly[13].Kpad, 512, ly[13].bd, ly[13].Opad,
                bufA, 128, 0, ly[13].O, ly[13].Opad, ly[13].Kpad, 1.0f, stream);
    launch_gemm(2, false, bufA, 128, ly[14].wd, ly[14].Kpad,
                (long long)ly[14].Opad * ly[14].Kpad, 512, ly[14].bd, ly[14].Opad,
                bufB, 128, 0, ly[14].O, ly[14].Opad, ly[14].Kpad, 1.0f, stream);
    // M2: tanh, f32 -> d_out cols 260..262
    launch_gemm(3, true, bufB, 128, ly[15].wd, ly[15].Kpad,
                (long long)ly[15].Opad * ly[15].Kpad, 512, ly[15].bd, ly[15].Opad,
                out, 263, 260, ly[15].O, ly[15].Opad, ly[15].Kpad, 1.0f, stream);
}
